// Attention_26740466385660
// MI455X (gfx1250) — matrix operand order repaired and checked by reading, not compiled, not run on hardware
//
#include <hip/hip_runtime.h>

// ---------- problem constants ----------
#define BATCH 2
#define NSEQ  3136        // 16*14*14
#define DIM   512
#define HEADS 8
#define HDIM  64
#define MROWS (BATCH * NSEQ)   // 6272 flat rows
// SCALE = HDIM^-0.5 = 0.125 (folded into Q at store time)

// ---------- vector types (trivial, union-safe) ----------
typedef __attribute__((ext_vector_type(16))) __bf16        bf16x16;
typedef __attribute__((ext_vector_type(8)))  float         f32x8;
typedef __attribute__((ext_vector_type(4)))  unsigned int  u32x4;

union BF {
    bf16x16 v;
    u32x4   q[2];
};

__device__ __forceinline__ f32x8 zero8() {
    f32x8 z = {0.f, 0.f, 0.f, 0.f, 0.f, 0.f, 0.f, 0.f};
    return z;
}

__device__ __forceinline__ f32x8 wmma_bf16(const BF& a, const BF& b, f32x8 c) {
    // D = A(16x32 bf16) x B(32x16 bf16) + C(16x16 f32)
    return __builtin_amdgcn_wmma_f32_16x16x32_bf16(
        false, a.v, false, b.v, (short)0, c, false, false);
}

// ---------- fp32 -> bf16 convert ----------
__global__ void cvt_bf16_kernel(const float* __restrict__ src,
                                __bf16* __restrict__ dst, int n) {
    int i = blockIdx.x * blockDim.x + threadIdx.x;
    if (i < n) dst[i] = (__bf16)src[i];
}

// ---------- QKV projection GEMM ----------
// X[6272,512] bf16 (row-major), W*[512,512] bf16 (row n = output feature).
// One wave computes a 16(M) x 64(N) tile; 64-col group == one head.
// z==0 -> Q [B,H,N,64] (scaled by 0.125), z==1 -> K [B,H,N,64],
// z==2 -> V stored transposed as VT [B,H,64,N].
__global__ __launch_bounds__(32) void qkv_gemm_kernel(
    const __bf16* __restrict__ X,
    const __bf16* __restrict__ WQ, const __bf16* __restrict__ WK,
    const __bf16* __restrict__ WV,
    __bf16* __restrict__ Q, __bf16* __restrict__ K, __bf16* __restrict__ VT)
{
    const int lane = threadIdx.x & 31;
    const int hl   = lane >> 4;       // half-wave select
    const int ln   = lane & 15;
    const int mt   = blockIdx.x;      // 0..391  (16-row tile)
    const int h    = blockIdx.y;      // 0..7    (head == 64-col group)
    const int w    = blockIdx.z;      // 0=Q 1=K 2=V
    const __bf16* Wm = (w == 0) ? WQ : ((w == 1) ? WK : WV);
    const int m0 = mt * 16;
    const int n0 = h * 64;

    f32x8 acc[4];
#pragma unroll
    for (int j = 0; j < 4; ++j) acc[j] = zero8();

    const __bf16* arow = X + (size_t)(m0 + ln) * DIM;
    for (int kk = 0; kk < DIM; kk += 32) {
        BF a;   // A fragment: lane row = ln, K-half selected by hl
        a.q[0] = *(const u32x4*)(arow + kk + hl * 16);
        a.q[1] = *(const u32x4*)(arow + kk + hl * 16 + 8);
#pragma unroll
        for (int j = 0; j < 4; ++j) {
            const __bf16* wrow = Wm + (size_t)(n0 + j * 16 + ln) * DIM + kk + hl * 16;
            BF b;   // B fragment: lane col = ln, K-half selected by hl
            b.q[0] = *(const u32x4*)(wrow);
            b.q[1] = *(const u32x4*)(wrow + 8);
            acc[j] = wmma_bf16(a, b, acc[j]);
        }
    }

    const int bidx  = m0 / NSEQ;      // 16-row tiles never cross batch (3136%16==0)
    const int qloc0 = m0 % NSEQ;
    if (w < 2) {
        __bf16* D = (w == 0) ? Q : K;
        const float scale = (w == 0) ? 0.125f : 1.0f;   // fold softmax scale into Q
#pragma unroll
        for (int j = 0; j < 4; ++j)
#pragma unroll
            for (int r = 0; r < 8; ++r) {
                int row = qloc0 + r + 8 * hl;
                size_t idx = ((((size_t)bidx * HEADS + h) * NSEQ) + row) * HDIM
                             + j * 16 + ln;
                D[idx] = (__bf16)(acc[j][r] * scale);
            }
    } else {
#pragma unroll
        for (int j = 0; j < 4; ++j)
#pragma unroll
            for (int r = 0; r < 8; ++r) {
                int row = qloc0 + r + 8 * hl;
                int col = j * 16 + ln;
                size_t idx = ((((size_t)bidx * HEADS + h) * HDIM) + col) * NSEQ + row;
                VT[idx] = (__bf16)acc[j][r];
            }
    }
}

// ---------- flash attention ----------
// One wave handles 16 query rows of one (b,h); online softmax over 98 chunks
// of 32 keys. 4 waves / block, each with a private 16x32 bf16 LDS tile for
// the S->P (C-layout -> A-layout) transpose.
__global__ __launch_bounds__(128) void attn_kernel(
    const __bf16* __restrict__ Q,   // [B,H,N,64] (pre-scaled)
    const __bf16* __restrict__ K,   // [B,H,N,64]
    const __bf16* __restrict__ VT,  // [B,H,64,N]
    __bf16* __restrict__ Aout)      // [B*N, 512]
{
    __shared__ __align__(16) __bf16 lds[4][16][32];

    const int tid  = threadIdx.x;
    const int wid  = tid >> 5;
    const int lane = tid & 31;
    const int hl   = lane >> 4;
    const int ln   = lane & 15;
    const int qt   = blockIdx.x * 4 + wid;    // 0..195
    const int bh   = blockIdx.y;              // 0..15
    const int b    = bh >> 3, h = bh & 7;
    const int m0   = qt * 16;                 // local query row base

    const __bf16* Qb = Q  + (size_t)bh * NSEQ * HDIM;
    const __bf16* Kb = K  + (size_t)bh * NSEQ * HDIM;
    const __bf16* Vb = VT + (size_t)bh * HDIM * NSEQ;

    // Q A-fragments for K-dim 0..31 and 32..63
    BF qa0, qa1;
    {
        const __bf16* qrow = Qb + (size_t)(m0 + ln) * HDIM;
        qa0.q[0] = *(const u32x4*)(qrow + hl * 16);
        qa0.q[1] = *(const u32x4*)(qrow + hl * 16 + 8);
        qa1.q[0] = *(const u32x4*)(qrow + 32 + hl * 16);
        qa1.q[1] = *(const u32x4*)(qrow + 32 + hl * 16 + 8);
    }

    f32x8 o[4];
#pragma unroll
    for (int j = 0; j < 4; ++j) o[j] = zero8();
    float mrow[8], lrow[8];
#pragma unroll
    for (int r = 0; r < 8; ++r) { mrow[r] = -1e30f; lrow[r] = 0.f; }

    for (int c0 = 0; c0 < NSEQ; c0 += 32) {
        // ---- S = Q x K^T for 32 keys (two 16x16 tiles) ----
        f32x8 s_lo = zero8(), s_hi = zero8();
        {
            const __bf16* krow = Kb + (size_t)(c0 + ln) * HDIM;        // keys c0..c0+15
            BF kb;
            kb.q[0] = *(const u32x4*)(krow + hl * 16);
            kb.q[1] = *(const u32x4*)(krow + hl * 16 + 8);
            s_lo = wmma_bf16(qa0, kb, s_lo);
            kb.q[0] = *(const u32x4*)(krow + 32 + hl * 16);
            kb.q[1] = *(const u32x4*)(krow + 32 + hl * 16 + 8);
            s_lo = wmma_bf16(qa1, kb, s_lo);

            const __bf16* krow2 = krow + 16 * HDIM;                    // keys +16
            kb.q[0] = *(const u32x4*)(krow2 + hl * 16);
            kb.q[1] = *(const u32x4*)(krow2 + hl * 16 + 8);
            s_hi = wmma_bf16(qa0, kb, s_hi);
            kb.q[0] = *(const u32x4*)(krow2 + 32 + hl * 16);
            kb.q[1] = *(const u32x4*)(krow2 + 32 + hl * 16 + 8);
            s_hi = wmma_bf16(qa1, kb, s_hi);
        }

        // ---- online softmax (row reduction across the 16-lane half) ----
        float mnew[8], alpha[8];
#pragma unroll
        for (int r = 0; r < 8; ++r) {
            float v = fmaxf(s_lo[r], s_hi[r]);
            v = fmaxf(v, __shfl_xor(v, 1));
            v = fmaxf(v, __shfl_xor(v, 2));
            v = fmaxf(v, __shfl_xor(v, 4));
            v = fmaxf(v, __shfl_xor(v, 8));
            mnew[r]  = fmaxf(mrow[r], v);
            alpha[r] = __expf(mrow[r] - mnew[r]);
            mrow[r]  = mnew[r];
        }
        f32x8 p_lo, p_hi;
#pragma unroll
        for (int r = 0; r < 8; ++r) {
            p_lo[r] = __expf(s_lo[r] - mnew[r]);
            p_hi[r] = __expf(s_hi[r] - mnew[r]);
            float sum = p_lo[r] + p_hi[r];
            sum += __shfl_xor(sum, 1);
            sum += __shfl_xor(sum, 2);
            sum += __shfl_xor(sum, 4);
            sum += __shfl_xor(sum, 8);
            lrow[r] = lrow[r] * alpha[r] + sum;
        }

        // ---- transpose P through LDS: C-layout -> A-layout ----
#pragma unroll
        for (int r = 0; r < 8; ++r) {
            lds[wid][r + 8 * hl][ln]      = (__bf16)p_lo[r];
            lds[wid][r + 8 * hl][16 + ln] = (__bf16)p_hi[r];
        }
        __syncthreads();
        BF pa;
        {
            const __bf16* prow = &lds[wid][ln][0];
            pa.q[0] = *(const u32x4*)(prow + hl * 16);
            pa.q[1] = *(const u32x4*)(prow + hl * 16 + 8);
        }
        __syncthreads();

        // ---- O = O*alpha + P x V ----
#pragma unroll
        for (int j = 0; j < 4; ++j)
#pragma unroll
            for (int r = 0; r < 8; ++r) o[j][r] *= alpha[r];
#pragma unroll
        for (int j = 0; j < 4; ++j) {
            const __bf16* vrow = Vb + (size_t)(j * 16 + ln) * NSEQ + c0 + hl * 16;
            BF vb;
            vb.q[0] = *(const u32x4*)(vrow);
            vb.q[1] = *(const u32x4*)(vrow + 8);
            o[j] = wmma_bf16(pa, vb, o[j]);
        }
    }

    // ---- normalize and store (head-concat layout for the proj GEMM) ----
#pragma unroll
    for (int r = 0; r < 8; ++r) lrow[r] = 1.0f / lrow[r];
#pragma unroll
    for (int j = 0; j < 4; ++j)
#pragma unroll
        for (int r = 0; r < 8; ++r) {
            int row = m0 + r + 8 * hl;
            size_t idx = ((size_t)b * NSEQ + row) * DIM + h * HDIM + j * 16 + ln;
            Aout[idx] = (__bf16)(o[j][r] * lrow[r]);
        }
}

// ---------- output projection: Out = A @ Wp^T (fp32 out) ----------
__global__ __launch_bounds__(32) void proj_gemm_kernel(
    const __bf16* __restrict__ A,   // [6272,512] bf16
    const __bf16* __restrict__ W,   // [512,512] bf16
    float* __restrict__ Out)        // [6272,512] fp32
{
    const int lane = threadIdx.x & 31;
    const int hl   = lane >> 4;
    const int ln   = lane & 15;
    const int m0   = blockIdx.x * 16;
    const int n0   = blockIdx.y * 64;

    f32x8 acc[4];
#pragma unroll
    for (int j = 0; j < 4; ++j) acc[j] = zero8();

    const __bf16* arow = A + (size_t)(m0 + ln) * DIM;
    for (int kk = 0; kk < DIM; kk += 32) {
        BF a;
        a.q[0] = *(const u32x4*)(arow + kk + hl * 16);
        a.q[1] = *(const u32x4*)(arow + kk + hl * 16 + 8);
#pragma unroll
        for (int j = 0; j < 4; ++j) {
            const __bf16* wrow = W + (size_t)(n0 + j * 16 + ln) * DIM + kk + hl * 16;
            BF b;
            b.q[0] = *(const u32x4*)(wrow);
            b.q[1] = *(const u32x4*)(wrow + 8);
            acc[j] = wmma_bf16(a, b, acc[j]);
        }
    }
#pragma unroll
    for (int j = 0; j < 4; ++j)
#pragma unroll
        for (int r = 0; r < 8; ++r) {
            size_t idx = (size_t)(m0 + r + 8 * hl) * DIM + n0 + j * 16 + ln;
            Out[idx] = acc[j][r];
        }
}

// ---------- host launcher ----------
extern "C" void kernel_launch(void* const* d_in, const int* in_sizes, int n_in,
                              void* d_out, int out_size, void* d_ws, size_t ws_size,
                              hipStream_t stream) {
    const float* x  = (const float*)d_in[0];
    const float* wq = (const float*)d_in[1];
    const float* wk = (const float*)d_in[2];
    const float* wv = (const float*)d_in[3];
    const float* wp = (const float*)d_in[4];
    float* out = (float*)d_out;

    // carve workspace (≈34 MB total)
    char* ws = (char*)d_ws;
    size_t off = 0;
    auto alloc = [&](size_t bytes) -> void* {
        off = (off + 255) & ~(size_t)255;
        void* p = ws + off;
        off += bytes;
        return p;
    };
    const size_t nX = (size_t)MROWS * DIM;       // 3,211,264
    const size_t nW = (size_t)DIM * DIM;         // 262,144
    const size_t nH = (size_t)BATCH * HEADS * NSEQ * HDIM; // == nX

    __bf16* xb  = (__bf16*)alloc(nX * 2);
    __bf16* wqb = (__bf16*)alloc(nW * 2);
    __bf16* wkb = (__bf16*)alloc(nW * 2);
    __bf16* wvb = (__bf16*)alloc(nW * 2);
    __bf16* wpb = (__bf16*)alloc(nW * 2);
    __bf16* Qb  = (__bf16*)alloc(nH * 2);
    __bf16* Kb  = (__bf16*)alloc(nH * 2);
    __bf16* Vtb = (__bf16*)alloc(nH * 2);
    __bf16* Ab  = (__bf16*)alloc(nX * 2);
    (void)ws_size; (void)in_sizes; (void)n_in; (void)out_size;

    // 1) fp32 -> bf16
    cvt_bf16_kernel<<<(int)((nX + 255) / 256), 256, 0, stream>>>(x,  xb,  (int)nX);
    cvt_bf16_kernel<<<(int)((nW + 255) / 256), 256, 0, stream>>>(wq, wqb, (int)nW);
    cvt_bf16_kernel<<<(int)((nW + 255) / 256), 256, 0, stream>>>(wk, wkb, (int)nW);
    cvt_bf16_kernel<<<(int)((nW + 255) / 256), 256, 0, stream>>>(wv, wvb, (int)nW);
    cvt_bf16_kernel<<<(int)((nW + 255) / 256), 256, 0, stream>>>(wp, wpb, (int)nW);

    // 2) QKV projections (Q pre-scaled, V transposed)
    qkv_gemm_kernel<<<dim3(MROWS / 16, HEADS, 3), 32, 0, stream>>>(
        xb, wqb, wkb, wvb, Qb, Kb, Vtb);

    // 3) flash attention
    attn_kernel<<<dim3(NSEQ / 16 / 4, BATCH * HEADS), 128, 0, stream>>>(
        Qb, Kb, Vtb, Ab);

    // 4) output projection
    proj_gemm_kernel<<<dim3(MROWS / 16, DIM / 64), 32, 0, stream>>>(Ab, wpb, out);
}
